// RecurrentMoE_49838800502874
// MI455X (gfx1250) — compile-verified, matrix operand order checked by reading, not run on hardware
//
#include <hip/hip_runtime.h>
#include <hip/hip_bf16.h>

// ---------------------------------------------------------------------------
// Types for CDNA5 WMMA (wave32, v_wmma_f32_16x16x32_bf16)
// ---------------------------------------------------------------------------
typedef __attribute__((ext_vector_type(16))) __bf16 v16bf;
typedef __attribute__((ext_vector_type(4)))  __bf16 v4bf;
typedef __attribute__((ext_vector_type(8)))  float  v8f;

#define CEILDIV(a, b) (((a) + (b) - 1) / (b))

// Model constants
#define Dc   1024
#define Hc   8
#define HDc  128
#define Sc   16
#define Ec   16
#define Kc   2
#define KRc  4
#define KWc  2
#define Bc   32
#define Tc   512
#define LKP  544   // padded Lk (multiple of 32, >= 528)

// ---------------------------------------------------------------------------
// Weight transpose + convert:  W (K,N) f32  ->  Wt (N,K) bf16
// LDS 32x32 tile transpose; both global accesses coalesced. blockIdx.z = batch.
// ---------------------------------------------------------------------------
__global__ __launch_bounds__(256) void k_wt(const float* __restrict__ W,
                                            __bf16* __restrict__ Wt, int K, int N)
{
    __shared__ float tile[32][33];
    const float* Wz = W + (size_t)blockIdx.z * K * N;
    __bf16*      Tz = Wt + (size_t)blockIdx.z * K * N;
    const int k0 = blockIdx.x * 32, n0 = blockIdx.y * 32;
    const int tid = threadIdx.x;
    #pragma unroll
    for (int i = 0; i < 4; ++i) {
        const int e = tid + i * 256;
        const int kr = e >> 5, nc = e & 31;
        float v = 0.f;
        if (k0 + kr < K && n0 + nc < N) v = Wz[(size_t)(k0 + kr) * N + n0 + nc];
        tile[kr][nc] = v;
    }
    __syncthreads();
    #pragma unroll
    for (int i = 0; i < 4; ++i) {
        const int e = tid + i * 256;
        const int nr = e >> 5, kc = e & 31;
        if (n0 + nr < N && k0 + kc < K)
            Tz[(size_t)(n0 + nr) * K + k0 + kc] = (__bf16)tile[kc][nr];
    }
}

// ---------------------------------------------------------------------------
// Batched WMMA GEMM:  C = epilogue(A @ W + bias)
//   A: (M,K) f32 row stride lda;  Wt: (N,K) bf16 pre-transposed (B fragments
//   loaded DIRECTLY from global, one 32B load per lane).  A double-buffered
//   through LDS (f32->bf16), one barrier per K-step, next tile prefetched to
//   registers before the WMMAs.  Edges handled by index clamping (garbage only
//   reaches discarded rows/cols).  Block tile 64(M) x 128(N), 8 waves, each
//   wave a 32x32 tile = 4 WMMAs per K-step.
//   Outputs: C (f32, optional) and/or Cbf (bf16, optional).
// ---------------------------------------------------------------------------
#define GTM 64
#define GTN 128
#define GTK 32

__global__ __launch_bounds__(256) void k_gemm(
    const float* __restrict__ A, int lda,
    const __bf16* __restrict__ Wt,
    const float* __restrict__ bias,
    const float* __restrict__ res,
    const float* __restrict__ pe, int pe_mod,
    float* __restrict__ C,
    __bf16* __restrict__ Cbf,
    int M, int N, int K, int act,
    long long aStride, int aDiv,
    const int* __restrict__ wIndex, long long wStride, long long bStride,
    long long cStride)
{
    __shared__ __align__(32) __bf16 As[2][GTM][GTK];   // 2 x 4KB

    const int z = blockIdx.z;
    const float* Az = A + (size_t)(z / aDiv) * (size_t)aStride;
    const __bf16* Wz = Wt;
    const float* Bz = bias;
    if (wIndex) {
        const int wi = wIndex[z];
        Wz = Wt + (size_t)wi * (size_t)wStride;
        if (bias) Bz = bias + (size_t)wi * (size_t)bStride;
    }
    float*  Cz   = C   ? C   + (size_t)z * (size_t)cStride : nullptr;
    __bf16* Cbz  = Cbf ? Cbf + (size_t)z * (size_t)cStride : nullptr;
    const float* Rz = res ? res + (size_t)z * (size_t)cStride : nullptr;

    const int bm = blockIdx.y * GTM, bn = blockIdx.x * GTN;
    const int tid = threadIdx.x, lane = tid & 31, wave = tid >> 5;
    const int wm = (wave >> 2) << 5;   // 0 or 32
    const int wn = (wave & 3) << 5;    // 0,32,64,96
    const int hf = lane >> 4, l16 = lane & 15;

    // A staging map: 2 float4 per thread (64 rows x 8 float4)
    int ar[2], ac[2], agr[2];
    #pragma unroll
    for (int i = 0; i < 2; ++i) {
        const int e = tid + i * 256;
        ar[i] = e >> 3; ac[i] = (e & 7) * 4;
        const int g = bm + ar[i];
        agr[i] = g < M ? g : (M - 1);          // clamp: pollutes only discarded rows
    }
    // B columns (clamped)
    const int n0s = bn + wn + l16;
    const int n1s = n0s + 16;
    const size_t n0c = (size_t)(n0s < N ? n0s : N - 1);
    const size_t n1c = (size_t)(n1s < N ? n1s : N - 1);

    // prologue: first A tile into registers
    float4 fa[2];
    #pragma unroll
    for (int i = 0; i < 2; ++i) fa[i] = *(const float4*)&Az[(size_t)agr[i] * lda + ac[i]];

    v8f a00 = {}, a01 = {}, a10 = {}, a11 = {};
    int cur = 0;
    for (int k0 = 0; k0 < K; k0 += GTK) {
        // convert + store staged A tile (packed b64 stores)
        #pragma unroll
        for (int i = 0; i < 2; ++i) {
            v4bf p = { (__bf16)fa[i].x, (__bf16)fa[i].y, (__bf16)fa[i].z, (__bf16)fa[i].w };
            *(v4bf*)&As[cur][ar[i]][ac[i]] = p;
        }
        // B fragments straight from global bf16 (32B per lane)
        const v16bf b0 = *(const v16bf*)&Wz[n0c * K + k0 + hf * 16];
        const v16bf b1 = *(const v16bf*)&Wz[n1c * K + k0 + hf * 16];
        __syncthreads();
        // prefetch next A tile into registers (overlaps with WMMAs below)
        if (k0 + GTK < K) {
            #pragma unroll
            for (int i = 0; i < 2; ++i)
                fa[i] = *(const float4*)&Az[(size_t)agr[i] * lda + k0 + GTK + ac[i]];
        }
        const v16bf a0 = *(const v16bf*)&As[cur][wm + l16][hf * 16];
        const v16bf a1 = *(const v16bf*)&As[cur][wm + 16 + l16][hf * 16];
        a00 = __builtin_amdgcn_wmma_f32_16x16x32_bf16(false, a0, false, b0, (short)0, a00, false, false);
        a01 = __builtin_amdgcn_wmma_f32_16x16x32_bf16(false, a0, false, b1, (short)0, a01, false, false);
        a10 = __builtin_amdgcn_wmma_f32_16x16x32_bf16(false, a1, false, b0, (short)0, a10, false, false);
        a11 = __builtin_amdgcn_wmma_f32_16x16x32_bf16(false, a1, false, b1, (short)0, a11, false, false);
        cur ^= 1;
    }

    auto epi = [&](v8f acc, int ms, int n) {
        if (n >= N) return;
        #pragma unroll
        for (int i = 0; i < 8; ++i) {
            const int m = bm + ms + hf * 8 + i;
            if (m >= M) continue;
            float v = acc[i];
            if (bias) v += Bz[n];
            if (pe)   v += pe[(size_t)(m % pe_mod) * N + n];
            if (act == 1)      v = 0.5f * v * (1.0f + erff(v * 0.70710678118654752f));
            else if (act == 2) v = fmaxf(v, 0.f);
            if (Rz) v += Rz[(size_t)m * N + n];
            if (Cz)  Cz[(size_t)m * N + n] = v;
            if (Cbz) Cbz[(size_t)m * N + n] = (__bf16)v;
        }
    };
    epi(a00, wm,      n0s);
    epi(a01, wm,      n1s);
    epi(a10, wm + 16, n0s);
    epi(a11, wm + 16, n1s);
}

// ---------------------------------------------------------------------------
// Fused attention: one wave per (b, h, 16-query tile).
// scores = (Q Kt)*scale -> softmax in LDS -> O = P V, all via bf16 WMMA.
//   qbf: (B, Lq, D) bf16; kbf: (B, Lk, D) bf16; vtbf: (B*H, HD, LKP) bf16.
// ---------------------------------------------------------------------------
__global__ __launch_bounds__(32) void k_attn(
    const __bf16* __restrict__ qbf, const __bf16* __restrict__ kbf,
    const __bf16* __restrict__ vtbf, float* __restrict__ outp,
    int Hn, int Lq, int Lk, float scale)
{
    __shared__ __align__(32) float  sc[16][LKP];
    __shared__ __align__(32) __bf16 pb[16][LKP];

    const int b  = blockIdx.y / Hn;
    const int h  = blockIdx.y % Hn;
    const int q0 = blockIdx.x * 16;
    const int lane = threadIdx.x & 31;
    const int hf  = lane >> 4;
    const int l16 = lane & 15;

    // Q fragments: HD=128 -> 4 K-steps of 32
    v16bf qF[4];
    const int qrow = q0 + l16;
    #pragma unroll
    for (int s = 0; s < 4; ++s) {
        v16bf t = {};
        if (qrow < Lq)
            t = *(const v16bf*)&qbf[((size_t)(b * Lq + qrow)) * Dc + h * HDc + s * 32 + hf * 16];
        qF[s] = t;
    }

    // scores, 16 keys per chunk
    for (int c = 0; c < LKP / 16; ++c) {
        v8f acc = {};
        const int kpos = c * 16 + l16;
        #pragma unroll
        for (int s = 0; s < 4; ++s) {
            v16bf kF = {};
            if (kpos < Lk)
                kF = *(const v16bf*)&kbf[((size_t)(b * Lk + kpos)) * Dc + h * HDc + s * 32 + hf * 16];
            acc = __builtin_amdgcn_wmma_f32_16x16x32_bf16(false, qF[s], false, kF,
                                                          (short)0, acc, false, false);
        }
        #pragma unroll
        for (int i = 0; i < 8; ++i)
            sc[hf * 8 + i][c * 16 + l16] = acc[i] * scale;
    }
    __syncthreads();

    // softmax: 2 lanes per row (lane r and r+16), combine via shfl_xor(16)
    {
        const int r = l16;
        float mx = -3.0e38f;
        for (int j = hf; j < Lk; j += 2) mx = fmaxf(mx, sc[r][j]);
        mx = fmaxf(mx, __shfl_xor(mx, 16, 32));
        float sum = 0.f;
        for (int j = hf; j < Lk; j += 2) { const float e = __expf(sc[r][j] - mx); sc[r][j] = e; sum += e; }
        sum += __shfl_xor(sum, 16, 32);
        const float inv = 1.0f / sum;
        for (int j = hf; j < Lk; j += 2)       pb[r][j] = (__bf16)(sc[r][j] * inv);
        for (int j = Lk + hf; j < LKP; j += 2) pb[r][j] = (__bf16)0.f;
    }
    __syncthreads();

    // O = P @ V : K-dim = LKP (17 steps), N-dim = 128 (8 tiles)
    v8f o[8];
    #pragma unroll
    for (int d = 0; d < 8; ++d) o[d] = (v8f){};
    for (int s = 0; s < LKP / 32; ++s) {
        const v16bf pF = *(const v16bf*)&pb[l16][s * 32 + hf * 16];
        #pragma unroll
        for (int d = 0; d < 8; ++d) {
            const v16bf vF = *(const v16bf*)&vtbf[((size_t)blockIdx.y * HDc + d * 16 + l16) * LKP
                                                  + s * 32 + hf * 16];
            o[d] = __builtin_amdgcn_wmma_f32_16x16x32_bf16(false, pF, false, vF,
                                                           (short)0, o[d], false, false);
        }
    }
    #pragma unroll
    for (int d = 0; d < 8; ++d)
        #pragma unroll
        for (int i = 0; i < 8; ++i) {
            const int m = q0 + hf * 8 + i;
            if (m < Lq)
                outp[((size_t)(b * Lq + m)) * Dc + h * HDc + d * 16 + l16] = o[d][i];
        }
}

// ---------------------------------------------------------------------------
// Elementwise / reduction helpers
// ---------------------------------------------------------------------------
__global__ __launch_bounds__(256) void k_pe(float* __restrict__ pe)
{
    const int i = blockIdx.x * 256 + threadIdx.x;
    if (i >= Tc * Dc) return;
    const int t = i >> 10, j = i & 1023;
    const int jj = (j < 512) ? j : j - 512;
    const float inv = powf((float)Sc, (float)jj / 512.0f);
    const float ang = (float)t / inv;
    pe[i] = (j < 512) ? sinf(ang) : cosf(ang);
}

__global__ __launch_bounds__(256) void k_layernorm(const float* __restrict__ in,
    const float* __restrict__ g, const float* __restrict__ bta, float* __restrict__ out)
{
    __shared__ float s1[256], s2[256];
    const int row = blockIdx.x, tid = threadIdx.x;
    const float* x = in + (size_t)row * Dc;
    float v[4], sum = 0.f, sq = 0.f;
    #pragma unroll
    for (int i = 0; i < 4; ++i) { v[i] = x[tid + i * 256]; sum += v[i]; sq += v[i] * v[i]; }
    s1[tid] = sum; s2[tid] = sq; __syncthreads();
    for (int off = 128; off; off >>= 1) {
        if (tid < off) { s1[tid] += s1[tid + off]; s2[tid] += s2[tid + off]; }
        __syncthreads();
    }
    const float mean = s1[0] * (1.0f / Dc);
    const float var  = s2[0] * (1.0f / Dc) - mean * mean;
    const float rstd = rsqrtf(var + 1e-5f);
    float* o = out + (size_t)row * Dc;
    #pragma unroll
    for (int i = 0; i < 4; ++i) { const int j = tid + i * 256; o[j] = (v[i] - mean) * rstd * g[j] + bta[j]; }
}

__global__ __launch_bounds__(256) void k_dot_rows(const float* __restrict__ in,
    const float* __restrict__ w, float* __restrict__ outp)
{
    __shared__ float s1[256];
    const int row = blockIdx.x, tid = threadIdx.x;
    const float* x = in + (size_t)row * Dc;
    float sum = 0.f;
    #pragma unroll
    for (int i = 0; i < 4; ++i) sum += x[tid + i * 256] * w[tid + i * 256];
    s1[tid] = sum; __syncthreads();
    for (int off = 128; off; off >>= 1) { if (tid < off) s1[tid] += s1[tid + off]; __syncthreads(); }
    if (tid == 0) outp[row] = s1[0];
}

__global__ void k_topk(const float* __restrict__ sc, int S_, int Kt, int dosm,
                       int* __restrict__ idx, float* __restrict__ val)
{
    if (threadIdx.x) return;
    const int b = blockIdx.x;
    const float* s = sc + (size_t)b * S_;
    float v[8]; int id[8]; bool used[64];
    for (int j = 0; j < S_; ++j) used[j] = false;
    for (int k = 0; k < Kt; ++k) {
        float best = -3.0e38f; int bi = 0;
        for (int j = 0; j < S_; ++j) if (!used[j] && s[j] > best) { best = s[j]; bi = j; }
        used[bi] = true; v[k] = best; id[k] = bi;
    }
    if (dosm) {
        const float mx = v[0]; float sum = 0.f; float e[8];
        for (int k = 0; k < Kt; ++k) { e[k] = __expf(v[k] - mx); sum += e[k]; }
        for (int k = 0; k < Kt; ++k) v[k] = e[k] / sum;
    }
    for (int k = 0; k < Kt; ++k) { idx[b * Kt + k] = id[k]; val[b * Kt + k] = v[k]; }
}

__global__ __launch_bounds__(256) void k_gather_lr(const float* __restrict__ latent,
    const int* __restrict__ ridx, const float* __restrict__ rv, float* __restrict__ lr, int n)
{
    const int i = blockIdx.x * 256 + threadIdx.x;
    if (i >= n) return;
    const int d = i & 1023, j = (i >> 10) & 3, b = i >> 12;
    const int s = ridx[b * KRc + j];
    lr[i] = latent[((size_t)(b * Sc + s)) * Dc + d] * rv[b * KRc + j];
}

__global__ __launch_bounds__(256) void k_concat(const float* __restrict__ s1, int L1,
    const float* __restrict__ s2, int L2, float* __restrict__ dst, int n)
{
    const int i = blockIdx.x * 256 + threadIdx.x;
    if (i >= n) return;
    const int d = i & 1023;
    const int r = (i >> 10) % (L1 + L2);
    const int b = (i >> 10) / (L1 + L2);
    dst[i] = (r < L1) ? s1[((size_t)(b * L1 + r)) * Dc + d]
                      : s2[((size_t)(b * L2 + (r - L1))) * Dc + d];
}

// vh (B, Lk, D) f32 -> vt (B*H, HD, LKP) bf16 via LDS 32x32 tile transpose
__global__ __launch_bounds__(256) void k_vt(const float* __restrict__ vh,
                                            __bf16* __restrict__ vt, int Lk)
{
    __shared__ float tile[32][33];
    const int l0 = blockIdx.x * 32;       // key position
    const int d0 = blockIdx.y * 32;       // head-dim
    const int bh = blockIdx.z;
    const int b = bh >> 3, h = bh & 7;
    const int tid = threadIdx.x;
    #pragma unroll
    for (int i = 0; i < 4; ++i) {
        const int e = tid + i * 256;
        const int lr = e >> 5, dc = e & 31;     // coalesced over d
        float v = 0.f;
        const int l = l0 + lr;
        if (l < Lk) v = vh[((size_t)(b * Lk + l)) * Dc + h * HDc + d0 + dc];
        tile[lr][dc] = v;
    }
    __syncthreads();
    #pragma unroll
    for (int i = 0; i < 4; ++i) {
        const int e = tid + i * 256;
        const int dr = e >> 5, lc = e & 31;     // coalesced over l
        vt[((size_t)bh * HDc + d0 + dr) * LKP + l0 + lc] = (__bf16)tile[lc][dr];
    }
}

__global__ __launch_bounds__(256) void k_mean_rows(const float* __restrict__ in,
                                                   float* __restrict__ out, int L, int n)
{
    const int i = blockIdx.x * 256 + threadIdx.x;
    if (i >= n) return;
    const int b = i >> 10, d = i & 1023;
    float s = 0.f;
    for (int t = 0; t < L; ++t) s += in[((size_t)(b * L + t)) * Dc + d];
    out[i] = s / (float)L;
}

__global__ __launch_bounds__(256) void k_moe_combine(float* __restrict__ lr,
    const float* __restrict__ yk, const float* __restrict__ gw, int n)
{
    const int i = blockIdx.x * 256 + threadIdx.x;
    if (i >= n) return;
    const int d = i & 1023, t = (i >> 10) & 3, b = i >> 12;
    float acc = lr[i];
    acc += yk[((size_t)(b * Kc + 0)) * (4 * Dc) + t * Dc + d] * gw[b * Kc + 0];
    acc += yk[((size_t)(b * Kc + 1)) * (4 * Dc) + t * Dc + d] * gw[b * Kc + 1];
    lr[i] = acc;
}

__global__ __launch_bounds__(256) void k_zero(float* __restrict__ p, int n)
{
    const int i = blockIdx.x * 256 + threadIdx.x;
    if (i < n) p[i] = 0.f;
}

__global__ __launch_bounds__(256) void k_scatter(const float* __restrict__ upd,
    const int* __restrict__ widx, const float* __restrict__ ww, float* __restrict__ state, int n)
{
    const int i = blockIdx.x * 256 + threadIdx.x;
    if (i >= n) return;
    const int d = i & 1023, j = (i >> 10) & 1, b = i >> 11;
    const int s = widx[b * KWc + j];   // distinct per b -> no conflicts
    state[((size_t)(b * Sc + s)) * Dc + d] = upd[(size_t)b * Dc + d] * ww[b * KWc + j];
}

// ---------------------------------------------------------------------------
// Host-side launch helpers
// ---------------------------------------------------------------------------
static void wtrans(hipStream_t s, const float* W, __bf16* Wt, int K, int N, int batches = 1)
{
    dim3 g(CEILDIV(K, 32), CEILDIV(N, 32), batches);
    k_wt<<<g, 256, 0, s>>>(W, Wt, K, N);
}

static void gemm(hipStream_t s, const float* A, int lda, const __bf16* Wt, const float* bias,
                 const float* res, const float* pe, int pe_mod, float* C, __bf16* Cbf,
                 int M, int N, int K, int act,
                 int batches = 1, long long aStride = 0, int aDiv = 1,
                 const int* wIndex = nullptr, long long wStride = 0, long long bStride = 0,
                 long long cStride = 0)
{
    dim3 g(CEILDIV(N, GTN), CEILDIV(M, GTM), batches);
    k_gemm<<<g, 256, 0, s>>>(A, lda, Wt, bias, res, pe, pe_mod, C, Cbf, M, N, K, act,
                             aStride, aDiv, wIndex, wStride, bStride, cStride);
}

// ---------------------------------------------------------------------------
// kernel_launch: full forward pipeline
// ---------------------------------------------------------------------------
extern "C" void kernel_launch(void* const* d_in, const int* in_sizes, int n_in,
                              void* d_out, int out_size, void* d_ws, size_t ws_size,
                              hipStream_t stream)
{
    (void)in_sizes; (void)n_in; (void)out_size;
    // ---- inputs (setup_inputs dict order) ----
    const float* x        = (const float*)d_in[0];
    const float* latent   = (const float*)d_in[1];   // (B, S, D)
    const float* s_emb_w  = (const float*)d_in[2];
    const float* s_emb_b  = (const float*)d_in[3];
    const float* o_emb_w  = (const float*)d_in[4];
    const float* o_emb_b  = (const float*)d_in[5];
    const float* s_wq = (const float*)d_in[6];
    const float* s_wk = (const float*)d_in[7];
    const float* s_wv = (const float*)d_in[8];
    const float* s_wo = (const float*)d_in[9];
    const float* s_bq = (const float*)d_in[10];
    const float* s_bk = (const float*)d_in[11];
    const float* s_bv = (const float*)d_in[12];
    const float* s_bo = (const float*)d_in[13];
    const float* s_fc1_w = (const float*)d_in[14];
    const float* s_fc1_b = (const float*)d_in[15];
    const float* s_fc2_w = (const float*)d_in[16];
    const float* s_fc2_b = (const float*)d_in[17];
    const float* o_wq = (const float*)d_in[18];
    const float* o_wk = (const float*)d_in[19];
    const float* o_wv = (const float*)d_in[20];
    const float* o_wo = (const float*)d_in[21];
    const float* o_bq = (const float*)d_in[22];
    const float* o_bk = (const float*)d_in[23];
    const float* o_bv = (const float*)d_in[24];
    const float* o_bo = (const float*)d_in[25];
    const float* o_fc1_w = (const float*)d_in[26];
    const float* o_fc1_b = (const float*)d_in[27];
    const float* o_fc2_w = (const float*)d_in[28];
    const float* o_fc2_b = (const float*)d_in[29];
    const float* ln_q_g   = (const float*)d_in[30];
    const float* ln_q_b   = (const float*)d_in[31];
    const float* ln_kv_g  = (const float*)d_in[32];
    const float* ln_kv_b  = (const float*)d_in[33];
    const float* ln_ffn_g = (const float*)d_in[34];
    const float* ln_ffn_b = (const float*)d_in[35];
    const float* ln_moe_g = (const float*)d_in[36];
    const float* ln_moe_b = (const float*)d_in[37];
    const float* ln_slot_g = (const float*)d_in[38];
    const float* ln_slot_b = (const float*)d_in[39];
    const float* oln_q_g   = (const float*)d_in[40];
    const float* oln_q_b   = (const float*)d_in[41];
    const float* oln_kv_g  = (const float*)d_in[42];
    const float* oln_kv_b  = (const float*)d_in[43];
    const float* oln_ffn_g = (const float*)d_in[44];
    const float* oln_ffn_b = (const float*)d_in[45];
    const float* gate_w = (const float*)d_in[46];
    const float* exp_W  = (const float*)d_in[47];
    const float* exp_b  = (const float*)d_in[48];
    const float* read_w = (const float*)d_in[49];
    const float* slot_w = (const float*)d_in[50];
    const float* out_proj_w = (const float*)d_in[51];
    const float* out_proj_b = (const float*)d_in[52];

    // ---- workspace arena (floats) ----
    float* base = (float*)d_ws;
    size_t off = 0;
    auto alloc = [&](size_t n) { float* p = base + off; off += (n + 63) & ~(size_t)63; return p; };
    const size_t DD = (size_t)Dc * Dc;

    // bf16 transposed weights (N,K)
    __bf16* WTsemb = (__bf16*)alloc(DD / 2);
    __bf16* WToemb = (__bf16*)alloc(DD / 2);
    __bf16* WTswq  = (__bf16*)alloc(DD / 2);
    __bf16* WTswk  = (__bf16*)alloc(DD / 2);
    __bf16* WTswv  = (__bf16*)alloc(DD / 2);
    __bf16* WTswo  = (__bf16*)alloc(DD / 2);
    __bf16* WTsf1  = (__bf16*)alloc(DD / 2);
    __bf16* WTsf2  = (__bf16*)alloc(DD / 2);
    __bf16* WTowq  = (__bf16*)alloc(DD / 2);
    __bf16* WTowk  = (__bf16*)alloc(DD / 2);
    __bf16* WTowv  = (__bf16*)alloc(DD / 2);
    __bf16* WTowo  = (__bf16*)alloc(DD / 2);
    __bf16* WTof1  = (__bf16*)alloc(DD / 2);
    __bf16* WTof2  = (__bf16*)alloc(DD / 2);
    __bf16* WTout  = (__bf16*)alloc(DD / 2);
    __bf16* WTgate = (__bf16*)alloc((size_t)Ec * Dc / 2);
    __bf16* WTexp  = (__bf16*)alloc((size_t)Ec * DD / 2);

    float* PE    = alloc((size_t)Tc * Dc);
    float* LR    = alloc((size_t)Bc * 4 * Dc);
    float* QLNS  = alloc((size_t)Bc * 4 * Dc);
    float* AOS   = alloc((size_t)Bc * 4 * Dc);
    float* T1    = alloc((size_t)Bc * 4 * Dc);
    float* T2    = alloc((size_t)Bc * 4 * Dc);
    float* POOL  = alloc((size_t)Bc * Dc);
    float* POOLLN= alloc((size_t)Bc * Dc);
    float* GL    = alloc((size_t)Bc * Ec);
    float* GW_   = alloc((size_t)Bc * Kc);
    int*   GIDX  = (int*)alloc((size_t)Bc * Kc);
    float* RS    = alloc((size_t)Bc * Sc);
    float* RV    = alloc((size_t)Bc * KRc);
    int*   RIDX  = (int*)alloc((size_t)Bc * KRc);
    float* SS    = alloc((size_t)Bc * Sc);
    float* WW    = alloc((size_t)Bc * KWc);
    int*   WIDX  = (int*)alloc((size_t)Bc * KWc);
    float* UPD   = alloc((size_t)Bc * Dc);
    float* LATLN = alloc((size_t)Bc * Sc * Dc);
    float* YK    = alloc((size_t)Bc * Kc * 4 * Dc);
    __bf16* QBFS = (__bf16*)alloc((size_t)Bc * 4 * Dc / 2);
    float* X1    = alloc((size_t)Bc * Tc * Dc);
    float* LO    = alloc((size_t)Bc * Tc * Dc);
    float* QH    = alloc((size_t)Bc * Tc * Dc);      // fc1 hidden (f32)
    float* AO    = alloc((size_t)Bc * Tc * Dc);
    float* KV    = alloc((size_t)Bc * 528 * Dc);
    float* VH    = alloc((size_t)Bc * 528 * Dc);
    __bf16* QBF  = (__bf16*)alloc((size_t)Bc * Tc * Dc / 2);
    __bf16* KBF  = (__bf16*)alloc((size_t)Bc * 528 * Dc / 2);
    __bf16* VBF  = (__bf16*)alloc((size_t)Bc * Hc * HDc * LKP / 2);
    if (off * sizeof(float) > ws_size) return;       // not enough scratch

    float* yOut     = (float*)d_out;             // (B, D)
    float* stateOut = (float*)d_out + Bc * Dc;   // (B, S, D)
    const float scale = 0.08838834764831845f;    // 1/sqrt(HD)

    // ---------------- Phase 0: weight prep + positional encoding ----------------
    wtrans(stream, s_emb_w, WTsemb, Dc, Dc);
    wtrans(stream, o_emb_w, WToemb, Dc, Dc);
    wtrans(stream, s_wq, WTswq, Dc, Dc);
    wtrans(stream, s_wk, WTswk, Dc, Dc);
    wtrans(stream, s_wv, WTswv, Dc, Dc);
    wtrans(stream, s_wo, WTswo, Dc, Dc);
    wtrans(stream, s_fc1_w, WTsf1, Dc, Dc);
    wtrans(stream, s_fc2_w, WTsf2, Dc, Dc);
    wtrans(stream, o_wq, WTowq, Dc, Dc);
    wtrans(stream, o_wk, WTowk, Dc, Dc);
    wtrans(stream, o_wv, WTowv, Dc, Dc);
    wtrans(stream, o_wo, WTowo, Dc, Dc);
    wtrans(stream, o_fc1_w, WTof1, Dc, Dc);
    wtrans(stream, o_fc2_w, WTof2, Dc, Dc);
    wtrans(stream, out_proj_w, WTout, Dc, Dc);
    wtrans(stream, gate_w, WTgate, Dc, Ec);
    wtrans(stream, exp_W, WTexp, Dc, Dc, Ec);
    k_pe<<<CEILDIV(Tc * Dc, 256), 256, 0, stream>>>(PE);

    // ---------------- Phase A: state mixed attention ----------------
    gemm(stream, x, Dc, WTsemb, s_emb_b, nullptr, PE, Tc, X1, nullptr, Bc * Tc, Dc, Dc, 0);
    k_dot_rows<<<Bc * Sc, 256, 0, stream>>>(latent, read_w, RS);
    k_topk<<<Bc, 32, 0, stream>>>(RS, Sc, KRc, 0, RIDX, RV);
    k_gather_lr<<<CEILDIV(Bc * 4 * Dc, 256), 256, 0, stream>>>(latent, RIDX, RV, LR, Bc * 4 * Dc);
    k_layernorm<<<Bc * 4, 256, 0, stream>>>(LR, ln_q_g, ln_q_b, QLNS);
    k_concat<<<CEILDIV(Bc * 516 * Dc, 256), 256, 0, stream>>>(LR, 4, X1, Tc, KV, Bc * 516 * Dc);
    k_layernorm<<<Bc * 516, 256, 0, stream>>>(KV, ln_kv_g, ln_kv_b, KV);
    // projections (q/k straight to bf16; v to f32 for transpose)
    gemm(stream, QLNS, Dc, WTswq, s_bq, nullptr, nullptr, 1, nullptr, QBFS, Bc * 4, Dc, Dc, 0);
    gemm(stream, KV, Dc, WTswk, s_bk, nullptr, nullptr, 1, nullptr, KBF, Bc * 516, Dc, Dc, 0);
    gemm(stream, KV, Dc, WTswv, s_bv, nullptr, nullptr, 1, VH, nullptr, Bc * 516, Dc, Dc, 0);
    {
        dim3 gv(CEILDIV(LKP, 32), HDc / 32, Bc * Hc);
        k_vt<<<gv, 256, 0, stream>>>(VH, VBF, 516);
        dim3 ga(1, Bc * Hc);
        k_attn<<<ga, 32, 0, stream>>>(QBFS, KBF, VBF, AOS, Hc, 4, 516, scale);
    }
    gemm(stream, AOS, Dc, WTswo, s_bo, LR, nullptr, 1, LR, nullptr, Bc * 4, Dc, Dc, 0);
    k_layernorm<<<Bc * 4, 256, 0, stream>>>(LR, ln_ffn_g, ln_ffn_b, T1);
    gemm(stream, T1, Dc, WTsf1, s_fc1_b, nullptr, nullptr, 1, T2, nullptr, Bc * 4, Dc, Dc, 1);
    gemm(stream, T2, Dc, WTsf2, s_fc2_b, LR, nullptr, 1, LR, nullptr, Bc * 4, Dc, Dc, 0);
    // MoE
    k_mean_rows<<<CEILDIV(Bc * Dc, 256), 256, 0, stream>>>(LR, POOL, 4, Bc * Dc);
    k_layernorm<<<Bc, 256, 0, stream>>>(POOL, ln_moe_g, ln_moe_b, POOLLN);
    gemm(stream, POOLLN, Dc, WTgate, nullptr, nullptr, nullptr, 1, GL, nullptr, Bc, Ec, Dc, 0);
    k_topk<<<Bc, 32, 0, stream>>>(GL, Ec, Kc, 1, GIDX, GW_);
    gemm(stream, LR, Dc, WTexp, exp_b, nullptr, nullptr, 1, YK, nullptr, 4, Dc, Dc, 2,
         Bc * Kc, (long long)4 * Dc, Kc, GIDX, (long long)DD, (long long)Dc,
         (long long)4 * Dc);
    k_moe_combine<<<CEILDIV(Bc * 4 * Dc, 256), 256, 0, stream>>>(LR, YK, GW_, Bc * 4 * Dc);
    // write gate -> state (directly into d_out)
    k_layernorm<<<Bc * Sc, 256, 0, stream>>>(latent, ln_slot_g, ln_slot_b, LATLN);
    k_dot_rows<<<Bc * Sc, 256, 0, stream>>>(LATLN, slot_w, SS);
    k_topk<<<Bc, 32, 0, stream>>>(SS, Sc, KWc, 1, WIDX, WW);
    k_mean_rows<<<CEILDIV(Bc * Dc, 256), 256, 0, stream>>>(LR, UPD, 4, Bc * Dc);
    k_zero<<<CEILDIV(Bc * Sc * Dc, 256), 256, 0, stream>>>(stateOut, Bc * Sc * Dc);
    k_scatter<<<CEILDIV(Bc * KWc * Dc, 256), 256, 0, stream>>>(UPD, WIDX, WW, stateOut, Bc * KWc * Dc);

    // ---------------- Phase B: output path ----------------
    gemm(stream, x, Dc, WToemb, o_emb_b, nullptr, PE, Tc, LO, nullptr, Bc * Tc, Dc, Dc, 0);
    k_layernorm<<<Bc * Tc, 256, 0, stream>>>(LO, oln_q_g, oln_q_b, X1);
    k_concat<<<CEILDIV(Bc * 528 * Dc, 256), 256, 0, stream>>>(LO, Tc, stateOut, Sc, KV, Bc * 528 * Dc);
    k_layernorm<<<Bc * 528, 256, 0, stream>>>(KV, oln_kv_g, oln_kv_b, KV);
    gemm(stream, X1, Dc, WTowq, o_bq, nullptr, nullptr, 1, nullptr, QBF, Bc * Tc, Dc, Dc, 0);
    gemm(stream, KV, Dc, WTowk, o_bk, nullptr, nullptr, 1, nullptr, KBF, Bc * 528, Dc, Dc, 0);
    gemm(stream, KV, Dc, WTowv, o_bv, nullptr, nullptr, 1, VH, nullptr, Bc * 528, Dc, Dc, 0);
    {
        dim3 gv(CEILDIV(LKP, 32), HDc / 32, Bc * Hc);
        k_vt<<<gv, 256, 0, stream>>>(VH, VBF, 528);
        dim3 ga(Tc / 16, Bc * Hc);
        k_attn<<<ga, 32, 0, stream>>>(QBF, KBF, VBF, AO, Hc, Tc, 528, scale);
    }
    gemm(stream, AO, Dc, WTowo, o_bo, LO, nullptr, 1, LO, nullptr, Bc * Tc, Dc, Dc, 0);
    k_layernorm<<<Bc * Tc, 256, 0, stream>>>(LO, oln_ffn_g, oln_ffn_b, X1);
    gemm(stream, X1, Dc, WTof1, o_fc1_b, nullptr, nullptr, 1, QH, nullptr, Bc * Tc, Dc, Dc, 1);
    gemm(stream, QH, Dc, WTof2, o_fc2_b, LO, nullptr, 1, LO, nullptr, Bc * Tc, Dc, Dc, 0);
    // y = lo[:, -1] @ out_proj + b   (strided A: one row per batch)
    gemm(stream, LO + (size_t)(Tc - 1) * Dc, Tc * Dc, WTout, out_proj_b,
         nullptr, nullptr, 1, yOut, nullptr, Bc, Dc, Dc, 0);
}
